// Switch_Layer_26259430048717
// MI455X (gfx1250) — 8-bit matrix operand order repaired and checked by reading, compiled, not run on hardware
//
#include <hip/hip_runtime.h>


#define BB 4
#define NN 1024
#define DD 256
#define HH 8
#define DHH 32
#define KHOPS 8
#define NEG_INF -1e9f

typedef __attribute__((ext_vector_type(2))) float v2f;
typedef __attribute__((ext_vector_type(8))) float v8f;
typedef __attribute__((ext_vector_type(4))) int   v4i;
typedef __attribute__((ext_vector_type(8))) int   v8i;

static __device__ __forceinline__ v8f wmma_f32_4(v2f a, v2f b, v8f c) {
  return __builtin_amdgcn_wmma_f32_16x16x4_f32(false, a, false, b, (short)0, c,
                                               false, false);
}

__global__ __launch_bounds__(32)
void gemm256_wmma(const float* __restrict__ A, const float* __restrict__ W,
                  float* __restrict__ C) {
  const int lane = threadIdx.x;
  const int half = lane >> 4, l16 = lane & 15;
  const int m0 = blockIdx.x * 16, n0 = blockIdx.y * 16;
  const float* arow = A + (size_t)(m0 + l16) * DD;
  v8f acc = {};
  for (int kk = 0; kk < DD; kk += 4) {
    __builtin_prefetch(&W[(kk + 8) * DD + n0], 0, 0);
    v2f a = *(const v2f*)(arow + kk + 2 * half);
    v2f b;
    b.x = W[(kk + 2 * half) * DD + n0 + l16];
    b.y = W[(kk + 2 * half + 1) * DD + n0 + l16];
    acc = wmma_f32_4(a, b, acc);
  }
#pragma unroll
  for (int r = 0; r < 8; ++r)
    C[(size_t)(m0 + r + 8 * half) * DD + n0 + l16] = acc[r];
}

__global__ __launch_bounds__(32)
void attn_kernel(const float* __restrict__ Q, const float* __restrict__ Km,
                 const float* __restrict__ V, const float* __restrict__ edge,
                 const int* __restrict__ adj, float* __restrict__ O) {
  __shared__ float S[16 * NN];
  const int lane = threadIdx.x, half = lane >> 4, l16 = lane & 15;
  const int mt = blockIdx.x, h = blockIdx.y, b = blockIdx.z;
  const int i0 = mt * 16;
  const float scale = 0.17677669529663687f;
  const float* qrow = Q + (size_t)(b * NN + i0 + l16) * DD + h * DHH;

  for (int jt = 0; jt < NN / 16; ++jt) {
    const int j0 = jt * 16;
    const float* krow = Km + (size_t)(b * NN + j0 + l16) * DD + h * DHH;
    v8f acc = {};
#pragma unroll
    for (int kk = 0; kk < DHH; kk += 4) {
      v2f a = *(const v2f*)(qrow + kk + 2 * half);
      v2f bb = *(const v2f*)(krow + kk + 2 * half);
      acc = wmma_f32_4(a, bb, acc);
    }
#pragma unroll
    for (int r = 0; r < 8; ++r) {
      const int li = r + 8 * half;
      const int i = i0 + li, j = j0 + l16;
      float s = acc[r] * scale +
                edge[((size_t)((b * HH + h) * NN + i)) * NN + j];
      s = (adj[((size_t)(b * NN + i)) * NN + j] > 0) ? s : NEG_INF;
      S[li * NN + j] = s;
    }
  }
  __syncthreads();

  for (int li = 0; li < 16; ++li) {
    float mx = NEG_INF;
    for (int j = lane; j < NN; j += 32) mx = fmaxf(mx, S[li * NN + j]);
    for (int off = 16; off > 0; off >>= 1)
      mx = fmaxf(mx, __shfl_xor(mx, off, 32));
    float sum = 0.f;
    for (int j = lane; j < NN; j += 32) {
      float e = __expf(S[li * NN + j] - mx);
      S[li * NN + j] = e;
      sum += e;
    }
    for (int off = 16; off > 0; off >>= 1) sum += __shfl_xor(sum, off, 32);
    const float inv = 1.0f / sum;
    for (int j = lane; j < NN; j += 32) S[li * NN + j] *= inv;
  }
  __syncthreads();

  v8f acc0 = {}, acc1 = {};
  for (int kk = 0; kk < NN; kk += 4) {
    v2f a = *(const v2f*)&S[l16 * NN + kk + 2 * half];
    const float* vb = V + (size_t)(b * NN + kk + 2 * half) * DD + h * DHH;
    v2f b0, b1;
    b0.x = vb[l16];       b0.y = vb[DD + l16];
    b1.x = vb[16 + l16];  b1.y = vb[DD + 16 + l16];
    acc0 = wmma_f32_4(a, b0, acc0);
    acc1 = wmma_f32_4(a, b1, acc1);
  }
#pragma unroll
  for (int r = 0; r < 8; ++r) {
    const int i = i0 + r + 8 * half;
    float* orow = O + (size_t)(b * NN + i) * DD + h * DHH;
    orow[l16] = acc0[r];
    orow[16 + l16] = acc1[r];
  }
}

__global__ void bfs_init(const int* __restrict__ adj,
                         unsigned char* __restrict__ aT,
                         unsigned char* __restrict__ cur,
                         unsigned char* __restrict__ dist,
                         unsigned char* __restrict__ expl) {
  size_t idx = (size_t)blockIdx.x * blockDim.x + threadIdx.x;
  if (idx >= (size_t)BB * NN * NN) return;
  const int j = (int)(idx % NN);
  const size_t t = idx / NN;
  const int i = (int)(t % NN);
  const int b = (int)(t / NN);
  const unsigned char v = (adj[idx] > 0) ? 1 : 0;
  cur[idx] = v;
  aT[((size_t)(b * NN + j)) * NN + i] = v;
  dist[idx] = v ? 0 : KHOPS;
  expl[idx] = v;
}

__global__ __launch_bounds__(32)
void bfs_hop(const unsigned char* __restrict__ cur,
             const unsigned char* __restrict__ aT,
             unsigned char* __restrict__ nxt,
             unsigned char* __restrict__ dist,
             unsigned char* __restrict__ expl, int hop) {
  const int lane = threadIdx.x, half = lane >> 4, l16 = lane & 15;
  const int m0 = blockIdx.x * 16, n0 = blockIdx.y * 16, b = blockIdx.z;
  const unsigned char* arow = cur + ((size_t)(b * NN + m0 + l16)) * NN;
  const unsigned char* brow = aT + ((size_t)(b * NN + n0 + l16)) * NN;
  v8i acc = {};
  for (int k0 = 0; k0 < NN; k0 += 64) {
    const int* ap = (const int*)(arow + k0) + 4 * half;
    v8i a;
    a[0] = ap[0];  a[1] = ap[1];  a[2] = ap[2];  a[3] = ap[3];
    a[4] = ap[8];  a[5] = ap[9];  a[6] = ap[10]; a[7] = ap[11];
    const unsigned char* bp = brow + k0 + 16 * half;
    const v4i q0 = *(const v4i*)(bp);
    const v4i q1 = *(const v4i*)(bp + 32);
    v8i bv;
    bv[0] = q0[0]; bv[1] = q0[1]; bv[2] = q0[2]; bv[3] = q0[3];
    bv[4] = q1[0]; bv[5] = q1[1]; bv[6] = q1[2]; bv[7] = q1[3];
    acc = __builtin_amdgcn_wmma_i32_16x16x64_iu8(false, a, false, bv, acc,
                                                 false, false);
  }
#pragma unroll
  for (int r = 0; r < 8; ++r) {
    const int i = m0 + r + 8 * half, j = n0 + l16;
    const size_t idx = ((size_t)(b * NN + i)) * NN + j;
    const unsigned char reach = (acc[r] > 0) ? 1 : 0;
    nxt[idx] = reach;
    if (reach && !expl[idx]) {
      expl[idx] = 1;
      dist[idx] = (unsigned char)hop;
    }
  }
}

__global__ __launch_bounds__(32)
void bias_fuse(const float* __restrict__ edge, const int* __restrict__ adj,
               const unsigned char* __restrict__ expl,
               const unsigned char* __restrict__ dist,
               const float* __restrict__ dw,
               float* __restrict__ bias_out, float* __restrict__ expl_out,
               float* __restrict__ F) {
  const int lane = threadIdx.x;
  const int i = blockIdx.x, h = blockIdx.y, b = blockIdx.z;
  const float* erow = edge + ((size_t)((b * HH + h) * NN + i)) * NN;
  const int* arow = adj + ((size_t)(b * NN + i)) * NN;
  const unsigned char* xrow = expl + ((size_t)(b * NN + i)) * NN;
  const unsigned char* drow = dist + ((size_t)(b * NN + i)) * NN;
  float* orow = bias_out + ((size_t)((b * HH + h) * NN + i)) * NN;
  float fsum = 0.f, dcount = 0.f;
  for (int j = lane; j < NN; j += 32) {
    const float e = erow[j];
    const unsigned char xp = xrow[j];
    int dc = drow[j];
    if (dc > KHOPS - 1) dc = KHOPS - 1;
    orow[j] = e + (xp ? dw[dc * HH + h] : 0.f);
    if (arow[j] > 0) {
      fsum += e;
      dcount += 1.f;
    }
    if (h == 0)
      expl_out[((size_t)(b * NN + i)) * NN + j] = xp ? 1.f : 0.f;
  }
  for (int off = 16; off > 0; off >>= 1) {
    fsum += __shfl_xor(fsum, off, 32);
    dcount += __shfl_xor(dcount, off, 32);
  }
  if (lane == 0) {
    const float deg = (dcount < 1.f) ? 1.f : dcount;
    F[((size_t)(b * NN + i)) * HH + h] = fsum / deg;
  }
}

__global__ __launch_bounds__(32)
void out_proj_fuse(const float* __restrict__ A, const float* __restrict__ W,
                   const float* __restrict__ x, const float* __restrict__ F,
                   const float* __restrict__ Wf, float* __restrict__ out) {
  const int lane = threadIdx.x, half = lane >> 4, l16 = lane & 15;
  const int m0 = blockIdx.x * 16, n0 = blockIdx.y * 16;
  const float* arow = A + (size_t)(m0 + l16) * DD;
  v8f acc = {};
  for (int kk = 0; kk < DD; kk += 4) {
    v2f a = *(const v2f*)(arow + kk + 2 * half);
    v2f b;
    b.x = W[(kk + 2 * half) * DD + n0 + l16];
    b.y = W[(kk + 2 * half + 1) * DD + n0 + l16];
    acc = wmma_f32_4(a, b, acc);
  }
#pragma unroll
  for (int r = 0; r < 8; ++r) {
    const int row = m0 + r + 8 * half, col = n0 + l16;
    float fuse = 0.f;
#pragma unroll
    for (int hh = 0; hh < HH; ++hh)
      fuse += F[(size_t)row * HH + hh] * Wf[hh * DD + col];
    out[(size_t)row * DD + col] = x[(size_t)row * DD + col] + acc[r] + fuse;
  }
}

extern "C" void kernel_launch(void* const* d_in, const int* in_sizes, int n_in,
                              void* d_out, int out_size, void* d_ws,
                              size_t ws_size, hipStream_t stream) {
  const float* x = (const float*)d_in[0];
  const int* adj = (const int*)d_in[1];
  const float* edge = (const float*)d_in[2];
  const float* Wq = (const float*)d_in[3];
  const float* Wk = (const float*)d_in[4];
  const float* Wv = (const float*)d_in[5];
  const float* Wo = (const float*)d_in[6];
  const float* Wfuse = (const float*)d_in[7];
  const float* dw = (const float*)d_in[8];

  char* ws = (char*)d_ws;
  const size_t MB = 1u << 20;
  float* Q = (float*)(ws + 0 * MB);
  float* Km = (float*)(ws + 4 * MB);
  float* V = (float*)(ws + 8 * MB);
  float* O = (float*)(ws + 12 * MB);
  float* F = (float*)(ws + 16 * MB);
  unsigned char* aT = (unsigned char*)(ws + 17 * MB);
  unsigned char* cur0 = (unsigned char*)(ws + 21 * MB);
  unsigned char* cur1 = (unsigned char*)(ws + 25 * MB);
  unsigned char* dist = (unsigned char*)(ws + 29 * MB);
  unsigned char* expl = (unsigned char*)(ws + 33 * MB);

  float* x2 = (float*)d_out;
  float* bias_out = x2 + (size_t)BB * NN * DD;
  float* expl_out = bias_out + (size_t)BB * HH * NN * NN;

  dim3 tgemm(BB * NN / 16, DD / 16);
  gemm256_wmma<<<tgemm, 32, 0, stream>>>(x, Wq, Q);
  gemm256_wmma<<<tgemm, 32, 0, stream>>>(x, Wk, Km);
  gemm256_wmma<<<tgemm, 32, 0, stream>>>(x, Wv, V);

  attn_kernel<<<dim3(NN / 16, HH, BB), 32, 0, stream>>>(Q, Km, V, edge, adj, O);

  {
    const size_t total = (size_t)BB * NN * NN;
    bfs_init<<<(unsigned)((total + 255) / 256), 256, 0, stream>>>(adj, aT, cur0,
                                                                  dist, expl);
    unsigned char* cin = cur0;
    unsigned char* cout = cur1;
    for (int hop = 1; hop < KHOPS; ++hop) {
      bfs_hop<<<dim3(NN / 16, NN / 16, BB), 32, 0, stream>>>(cin, aT, cout,
                                                             dist, expl, hop);
      unsigned char* t = cin;
      cin = cout;
      cout = t;
    }
  }

  bias_fuse<<<dim3(NN, HH, BB), 32, 0, stream>>>(edge, adj, expl, dist, dw,
                                                 bias_out, expl_out, F);

  out_proj_fuse<<<tgemm, 32, 0, stream>>>(O, Wo, x, F, Wfuse, x2);
}
